// FusionNeRF_3032246911660
// MI455X (gfx1250) — matrix operand order repaired and checked by reading, not compiled, not run on hardware
//
#include <hip/hip_runtime.h>

#define R_RAYS 2048
#define S_SAMP 128
#define NPTS   (R_RAYS * S_SAMP)
#define ROWS   64
#define HID    256

typedef _Float16 v16h __attribute__((ext_vector_type(16)));
typedef _Float16 h8   __attribute__((ext_vector_type(8)));
typedef float    v8f  __attribute__((ext_vector_type(8)));

union Frag { v16h v; h8 h[2]; };

__device__ __forceinline__ float sigmoidf(float x) {
    return 1.0f / (1.0f + __expf(-x));
}

// Cooperatively stage one 32x256 f16 weight k-slice (16 KB) into LDS using the
// CDNA5 async global->LDS path (ASYNCcnt-tracked). Layout in LDS:
//   dst[((part * 256) + col) * 8]  -- part = 16-byte sub-slice of the 64B column slice.
// 256 threads x 4 chunks = 1024 x 16B. 4 ASYNCcnt increments per wave per stage.
__device__ __forceinline__ void stage_w(const _Float16* __restrict__ Wt, int KPAD,
                                        int k0, _Float16* dst)
{
    const int t = threadIdx.x;              // column index 0..255
#pragma unroll
    for (int part = 0; part < 4; ++part) {
        const _Float16* g = Wt + t * KPAD + k0 + part * 8;          // 16B aligned
        unsigned ldsOff = (unsigned)(unsigned long long)(dst + (part * 256 + t) * 8);
        asm volatile("global_load_async_to_lds_b128 %0, %1, off"
                     :: "v"(ldsOff), "v"(g) : "memory");
    }
}

// C[64x256] = act( A[64xKPAD] @ Wt^T + b ).  A in LDS; B double-buffer staged into
// LDS via async copies overlapped with WMMA.  8 waves: wave (wid&3) -> 16-row tile,
// (wid>>2) -> 128-column half; 8 f32 accumulators per wave.
template<int KPAD, bool RELU>
__device__ __forceinline__ void gemm_layer(const _Float16* aIn,
                                           const _Float16* __restrict__ Wt,
                                           const float* __restrict__ bias,
                                           _Float16* aOut,
                                           _Float16* wstage /* 2 x 8192 halves */)
{
    constexpr int KSTEPS = KPAD / 32;
    const int lane  = threadIdx.x & 31;
    const int wid   = threadIdx.x >> 5;
    const int rbase = (wid & 3) * 16;
    const int cbase = (wid >> 2) * 128;
    const int sel   = lane >> 4;      // lane half selects K sub-block per WMMA layout
    const int mr    = rbase + (lane & 15);
    const int nlo   = lane & 15;

    v8f acc[8] = {};
    stage_w(Wt, KPAD, 0, wstage);                       // prologue: stage k-step 0
    for (int ks = 0; ks < KSTEPS; ++ks) {
        _Float16* cur = wstage + (ks & 1) * 8192;
        if (ks + 1 < KSTEPS) {
            stage_w(Wt, KPAD, (ks + 1) * 32, wstage + ((ks + 1) & 1) * 8192);
            asm volatile("s_wait_asynccnt 4" ::: "memory");  // stage(ks) done (in-order)
        } else {
            asm volatile("s_wait_asynccnt 0" ::: "memory");
        }
        __syncthreads();   // all waves' stage(ks) writes visible

        const int k0 = ks * 32;
        // A fragment (16x32 f16): elems 0..7 -> K = k0+sel*8.. ; 8..15 -> K = k0+16+sel*8..
        Frag a;
        a.h[0] = *reinterpret_cast<const h8*>(aIn + mr * KPAD + k0 + sel * 8);
        a.h[1] = *reinterpret_cast<const h8*>(aIn + mr * KPAD + k0 + 16 + sel * 8);
#pragma unroll
        for (int ti = 0; ti < 8; ++ti) {
            const int n = cbase + ti * 16 + nlo;
            // B fragment (32x16 f16): elems 0..7 -> K = sel*8.. ; 8..15 -> K = 16+sel*8.. for column n, from LDS
            Frag b;
            b.h[0] = *reinterpret_cast<const h8*>(cur + ((sel)     * 256 + n) * 8);
            b.h[1] = *reinterpret_cast<const h8*>(cur + ((2 + sel) * 256 + n) * 8);
            acc[ti] = __builtin_amdgcn_wmma_f32_16x16x32_f16(
                false, a.v, false, b.v, (short)0, acc[ti], false, false);
        }
        __syncthreads();   // all waves done reading cur before it is restaged
    }
    // Epilogue: C/D layout -> VGPR r, lane l: row = rbase + r + (l>>4)*8, col = cbase+ti*16+(l&15)
#pragma unroll
    for (int ti = 0; ti < 8; ++ti) {
        const int col = cbase + ti * 16 + nlo;
        const float bv = bias[col];
#pragma unroll
        for (int r = 0; r < 8; ++r) {
            const int row = rbase + r + sel * 8;
            float v = acc[ti][r] + bv;
            if (RELU) v = v > 0.0f ? v : 0.0f;
            aOut[row * HID + col] = (_Float16)v;
        }
    }
}

// Small output head: out[ROWS][OUTC] = act[ROWS][256] @ W[256][OUTC] + b
__device__ __forceinline__ void head_dot(const _Float16* act,
                                         const float* __restrict__ W,
                                         const float* __restrict__ b,
                                         float* out, int OUTC)
{
    for (int idx = threadIdx.x; idx < ROWS * OUTC; idx += 256) {
        int r = idx / OUTC, c = idx - r * OUTC;
        float s = b[c];
        for (int k = 0; k < HID; ++k)
            s += (float)act[r * HID + k] * W[k * OUTC + c];
        out[idx] = s;
    }
}

struct Params {
    const float *points, *dirs, *timev;
    const _Float16 *sW0t, *sW1t, *sW2t, *sW3t;
    const float *sb0, *sb1, *sb2, *sb3, *sW4, *sb4;
    const _Float16 *dW0t, *dW1t, *dW2t, *dW3t;
    const float *db0, *db1, *db2, *db3, *dW4, *db4;
    float *midSigma, *midBw, *midRgb;
};

__global__ __launch_bounds__(256) void nerf_fused(Params P)
{
    __shared__ alignas(16) _Float16 encS[ROWS * 96];    // static input, 90 padded to 96
    __shared__ alignas(16) _Float16 encD[ROWS * 128];   // dynamic input, 99 padded to 128
    __shared__ alignas(16) _Float16 actA[ROWS * HID];
    __shared__ alignas(16) _Float16 actB[ROWS * HID];
    __shared__ alignas(16) _Float16 wstage[2 * 32 * 256]; // async weight double-buffer
    __shared__ float headS[ROWS * 4];
    __shared__ float headD[ROWS * 5];

    const int t    = threadIdx.x;
    const int row0 = blockIdx.x * ROWS;

    // Positional encoding: one thread per point (torch order: raw, then per-band [sin*3, cos*3])
    if (t < ROWS) {
        const int gi = row0 + t;
        float p[3], d[3];
        p[0] = P.points[gi*3];  p[1] = P.points[gi*3+1];  p[2] = P.points[gi*3+2];
        d[0] = P.dirs[gi*3];    d[1] = P.dirs[gi*3+1];    d[2] = P.dirs[gi*3+2];
        float tm = P.timev[gi];
        _Float16* es = encS + t * 96;
        _Float16* ed = encD + t * 128;
        for (int c = 0; c < 3; ++c) { es[c] = (_Float16)p[c]; ed[c] = (_Float16)p[c]; }
        for (int l = 0; l < 10; ++l) {
            float f = (float)(1 << l);
            for (int c = 0; c < 3; ++c) {
                float sv = __sinf(f * p[c]), cv = __cosf(f * p[c]);
                es[3+6*l+c]   = (_Float16)sv;  ed[3+6*l+c]   = (_Float16)sv;
                es[3+6*l+3+c] = (_Float16)cv;  ed[3+6*l+3+c] = (_Float16)cv;
            }
        }
        for (int c = 0; c < 3; ++c) { es[63+c] = (_Float16)d[c]; ed[63+c] = (_Float16)d[c]; }
        for (int l = 0; l < 4; ++l) {
            float f = (float)(1 << l);
            for (int c = 0; c < 3; ++c) {
                float sv = __sinf(f * d[c]), cv = __cosf(f * d[c]);
                es[66+6*l+c]   = (_Float16)sv;  ed[66+6*l+c]   = (_Float16)sv;
                es[66+6*l+3+c] = (_Float16)cv;  ed[66+6*l+3+c] = (_Float16)cv;
            }
        }
        for (int c = 90; c < 96; ++c) es[c] = (_Float16)0.0f;
        ed[90] = (_Float16)tm;
        for (int l = 0; l < 4; ++l) {
            float f = (float)(1 << l);
            ed[91+2*l] = (_Float16)__sinf(f * tm);
            ed[92+2*l] = (_Float16)__cosf(f * tm);
        }
        for (int c = 99; c < 128; ++c) ed[c] = (_Float16)0.0f;
    }
    __syncthreads();

    // ---- static MLP, activations resident in LDS ----
    gemm_layer<96,  true>(encS, P.sW0t, P.sb0, actA, wstage); __syncthreads();
    gemm_layer<256, true>(actA, P.sW1t, P.sb1, actB, wstage); __syncthreads();
    gemm_layer<256, true>(actB, P.sW2t, P.sb2, actA, wstage); __syncthreads();
    gemm_layer<256, true>(actA, P.sW3t, P.sb3, actB, wstage); __syncthreads();
    head_dot(actB, P.sW4, P.sb4, headS, 4);            // reads actB before anyone rewrites it

    // ---- dynamic MLP ----
    gemm_layer<128, true>(encD, P.dW0t, P.db0, actA, wstage); __syncthreads();
    gemm_layer<256, true>(actA, P.dW1t, P.db1, actB, wstage); __syncthreads();
    gemm_layer<256, true>(actB, P.dW2t, P.db2, actA, wstage); __syncthreads();
    gemm_layer<256, true>(actA, P.dW3t, P.db3, actB, wstage); __syncthreads();
    head_dot(actB, P.dW4, P.db4, headD, 5);
    __syncthreads();

    // ---- blend heads, emit per-point sigma / rgb / blend weight ----
    if (t < ROWS) {
        const int gi = row0 + t;
        float bw  = sigmoidf(headD[t*5 + 4]);
        float sig = (1.0f - bw) * headS[t*4] + bw * headD[t*5];
        P.midSigma[gi] = sig;
        P.midBw[gi]    = bw;
        for (int c = 0; c < 3; ++c) {
            float rc = (1.0f - bw) * sigmoidf(headS[t*4 + 1 + c])
                     +         bw  * sigmoidf(headD[t*5 + 1 + c]);
            P.midRgb[gi*3 + c] = rc;
        }
    }
}

__global__ void convert_wt(const float* __restrict__ W, _Float16* __restrict__ Wt,
                           int Kreal, int KPAD, int NCOLS)
{
    int idx = blockIdx.x * blockDim.x + threadIdx.x;
    if (idx >= NCOLS * KPAD) return;
    int n = idx / KPAD, k = idx - n * KPAD;
    Wt[idx] = (k < Kreal) ? (_Float16)W[k * NCOLS + n] : (_Float16)0.0f;
}

__global__ void composite(const float* __restrict__ sigma, const float* __restrict__ bw,
                          const float* __restrict__ rgb, const float* __restrict__ z,
                          float* __restrict__ out)
{
    int ray = blockIdx.x * blockDim.x + threadIdx.x;
    if (ray >= R_RAYS) return;
    float* rgb_map = out;
    float* depth   = out + R_RAYS * 3;
    float* wts     = depth + R_RAYS;
    float* swts    = wts + NPTS;
    float* dwts    = swts + NPTS;
    float T = 1.0f, r0 = 0, r1 = 0, r2 = 0, dm = 0;
    for (int s = 0; s < S_SAMP; ++s) {
        int i = ray * S_SAMP + s;
        float zv    = z[i];
        float delta = (s < S_SAMP - 1) ? (z[i+1] - zv) : 1e10f;
        float alpha = 1.0f - __expf(-sigma[i] * delta);
        float w     = alpha * T;
        T *= (1.0f - alpha + 1e-10f);
        float b = bw[i];
        wts[i]  = w;
        swts[i] = (1.0f - b) * w;
        dwts[i] = b * w;
        r0 += w * rgb[i*3]; r1 += w * rgb[i*3+1]; r2 += w * rgb[i*3+2];
        dm += w * zv;
    }
    rgb_map[ray*3]   = r0;
    rgb_map[ray*3+1] = r1;
    rgb_map[ray*3+2] = r2;
    depth[ray]       = dm;
}

extern "C" void kernel_launch(void* const* d_in, const int* in_sizes, int n_in,
                              void* d_out, int out_size, void* d_ws, size_t ws_size,
                              hipStream_t stream)
{
    const float* points = (const float*)d_in[0];
    const float* dirs   = (const float*)d_in[1];
    const float* zvals  = (const float*)d_in[2];
    const float* timev  = (const float*)d_in[3];
    const float *sW[5], *sb[5], *dW[5], *db[5];
    for (int i = 0; i < 5; ++i) {                 // setup order: sW_i, sb_i, dW_i, db_i interleaved
        sW[i] = (const float*)d_in[4 + 4*i];
        sb[i] = (const float*)d_in[5 + 4*i];
        dW[i] = (const float*)d_in[6 + 4*i];
        db[i] = (const float*)d_in[7 + 4*i];
    }

    char* w = (char*)d_ws;
    auto alloc = [&](size_t b) { char* p = w; w += (b + 255) & ~(size_t)255; return p; };
    _Float16* sW0t = (_Float16*)alloc(256 * 96  * 2);
    _Float16* sW1t = (_Float16*)alloc(256 * 256 * 2);
    _Float16* sW2t = (_Float16*)alloc(256 * 256 * 2);
    _Float16* sW3t = (_Float16*)alloc(256 * 256 * 2);
    _Float16* dW0t = (_Float16*)alloc(256 * 128 * 2);
    _Float16* dW1t = (_Float16*)alloc(256 * 256 * 2);
    _Float16* dW2t = (_Float16*)alloc(256 * 256 * 2);
    _Float16* dW3t = (_Float16*)alloc(256 * 256 * 2);
    float* midSigma = (float*)alloc((size_t)NPTS * 4);
    float* midBw    = (float*)alloc((size_t)NPTS * 4);
    float* midRgb   = (float*)alloc((size_t)NPTS * 12);

    auto cvt = [&](const float* src, _Float16* dst, int kr, int kp) {
        int total = 256 * kp;
        convert_wt<<<(total + 255) / 256, 256, 0, stream>>>(src, dst, kr, kp, 256);
    };
    cvt(sW[0], sW0t, 90, 96);   cvt(sW[1], sW1t, 256, 256);
    cvt(sW[2], sW2t, 256, 256); cvt(sW[3], sW3t, 256, 256);
    cvt(dW[0], dW0t, 99, 128);  cvt(dW[1], dW1t, 256, 256);
    cvt(dW[2], dW2t, 256, 256); cvt(dW[3], dW3t, 256, 256);

    Params P;
    P.points = points; P.dirs = dirs; P.timev = timev;
    P.sW0t = sW0t; P.sW1t = sW1t; P.sW2t = sW2t; P.sW3t = sW3t;
    P.sb0 = sb[0]; P.sb1 = sb[1]; P.sb2 = sb[2]; P.sb3 = sb[3];
    P.sW4 = sW[4]; P.sb4 = sb[4];
    P.dW0t = dW0t; P.dW1t = dW1t; P.dW2t = dW2t; P.dW3t = dW3t;
    P.db0 = db[0]; P.db1 = db[1]; P.db2 = db[2]; P.db3 = db[3];
    P.dW4 = dW[4]; P.db4 = db[4];
    P.midSigma = midSigma; P.midBw = midBw; P.midRgb = midRgb;

    nerf_fused<<<NPTS / ROWS, 256, 0, stream>>>(P);
    composite<<<(R_RAYS + 255) / 256, 256, 0, stream>>>(midSigma, midBw, midRgb, zvals,
                                                        (float*)d_out);
}
